// MultiHeadAttention_48765058679009
// MI455X (gfx1250) — matrix operand order repaired and checked by reading, not compiled, not run on hardware
//
#include <hip/hip_runtime.h>

typedef __attribute__((ext_vector_type(16))) _Float16 v16h;
typedef __attribute__((ext_vector_type(8)))  _Float16 v8h;
typedef __attribute__((ext_vector_type(8)))  float    v8f;

#define B_    2
#define S_    2048
#define HID_  1024
#define NH_   16
#define DK_   64
#define M_TOT (B_ * S_)          // 4096
#define NEGV  (-1000000000.0f)
#define SCALE_ 0.125f            // 64^-0.5
#define QT_   2                  // q-tiles (16 rows each) per attention block

enum { MODE_HEADS = 0, MODE_HEADS_T = 1, MODE_F32 = 2 };

static __device__ __forceinline__ v16h cat8(v8h lo, v8h hi) {
  v16h r;
#pragma unroll
  for (int i = 0; i < 8; ++i) { r[i] = lo[i]; r[i + 8] = hi[i]; }
  return r;
}

static __device__ __forceinline__ v8f zero8() {
  v8f z;
#pragma unroll
  for (int i = 0; i < 8; ++i) z[i] = 0.0f;
  return z;
}

// ---------------------------------------------------------------------------
// f16 WMMA GEMM: out(4096x1024) = A(4096x1024) x Bw(1024x1024)
// Block: 128 threads (4 waves), tile 64x64, K-step 32.
// mode selects epilogue layout (per-head f16, per-head transposed f16, plain f32)
// ---------------------------------------------------------------------------
template <bool AF16>
__global__ void __launch_bounds__(128)
gemm_mha(const void* __restrict__ Aglob, const float* __restrict__ Bw,
         void* __restrict__ out, int mode, float outScale) {
  __shared__ _Float16 ldsA[64 * 48];   // [m][k], stride 48 halves (96B, 16B-mult)
  __shared__ _Float16 ldsB[64 * 48];   // transposed: [n][k]

  const int tid  = threadIdx.x;
  const int lane = tid & 31;
  const int w    = tid >> 5;
  const int l15  = lane & 15;
  const int hi   = lane >> 4;
  const int m0   = blockIdx.y * 64;
  const int n0   = blockIdx.x * 64;

  v8f acc[4];
#pragma unroll
  for (int j = 0; j < 4; ++j) acc[j] = zero8();

  const float*    Af = (const float*)Aglob;
  const _Float16* Ah = (const _Float16*)Aglob;

  for (int k0 = 0; k0 < HID_; k0 += 32) {
    // prefetch next K-tile (global_prefetch_b8)
    if (k0 + 32 < HID_) {
      const int pr = tid & 63;
      __builtin_prefetch(AF16 ? (const void*)&Ah[(m0 + pr) * HID_ + k0 + 32]
                              : (const void*)&Af[(m0 + pr) * HID_ + k0 + 32],
                         0, 1);
      __builtin_prefetch(&Bw[(k0 + 32 + (tid & 31)) * HID_ + n0], 0, 1);
    }
    __syncthreads();
    // stage A tile 64x32 (convert to f16), coalesced along k
#pragma unroll
    for (int i = tid; i < 64 * 32; i += 128) {
      const int row = i >> 5, col = i & 31;
      const int gi  = (m0 + row) * HID_ + (k0 + col);
      ldsA[row * 48 + col] = AF16 ? Ah[gi] : (_Float16)Af[gi];
    }
    // stage B tile 32x64 transposed -> [n][k], coalesced along n
#pragma unroll
    for (int i = tid; i < 32 * 64; i += 128) {
      const int kr = i >> 6, nc = i & 63;
      ldsB[nc * 48 + kr] = (_Float16)Bw[(k0 + kr) * HID_ + (n0 + nc)];
    }
    __syncthreads();

    // A fragment: row = 16w + l15; halves: K = 8*hi + h (h<8), 16 + 8*hi + (h-8)
    const _Float16* ap = &ldsA[((w << 4) + l15) * 48];
    const v16h a = cat8(*(const v8h*)(ap + 8 * hi),
                        *(const v8h*)(ap + 16 + 8 * hi));
#pragma unroll
    for (int j = 0; j < 4; ++j) {
      // B fragment: col = 16j + l15; halves: K = 8*hi + h (h<8), 16 + 8*hi + (h-8)
      const _Float16* bp = &ldsB[((j << 4) + l15) * 48];
      const v16h b = cat8(*(const v8h*)(bp + 8 * hi), *(const v8h*)(bp + 16 + 8 * hi));
      acc[j] = __builtin_amdgcn_wmma_f32_16x16x32_f16(
          false, a, false, b, (short)0, acc[j], false, false);
    }
  }

  // epilogue: C layout -> lane row m = r + 8*hi, col n = l15
#pragma unroll
  for (int j = 0; j < 4; ++j) {
#pragma unroll
    for (int r = 0; r < 8; ++r) {
      const int gm = m0 + (w << 4) + r + (hi << 3);
      const int gn = n0 + (j << 4) + l15;
      const float val = acc[j][r] * outScale;
      if (mode == MODE_F32) {
        ((float*)out)[(size_t)gm * HID_ + gn] = val;
      } else {
        const int b  = gm >> 11;       // / S_
        const int s  = gm & (S_ - 1);
        const int hd = gn >> 6;        // / DK_
        const int dk = gn & (DK_ - 1);
        const int bh = (b << 4) + hd;  // b*NH + head
        const size_t idx = (mode == MODE_HEADS)
            ? ((size_t)bh * S_ + s) * DK_ + dk       // [b,h,s,dk]
            : ((size_t)bh * DK_ + dk) * S_ + s;      // [b,h,dk,s]
        ((_Float16*)out)[idx] = (_Float16)val;
      }
    }
  }
}

// ---------------------------------------------------------------------------
// Flash attention: one wave per (b, head, 32-query block = 2 q-tiles).
// Q pre-scaled f16 [b,h,s,dk]; K f16 [b,h,s,dk]; V f16 transposed [b,h,dk,s].
// Processing 2 q-tiles per wave halves per-head K/V re-read traffic.
// ---------------------------------------------------------------------------
__global__ void __launch_bounds__(32)
attn_mha(const _Float16* __restrict__ Q, const _Float16* __restrict__ K,
         const _Float16* __restrict__ Vt, const int* __restrict__ mask,
         _Float16* __restrict__ O) {
  __shared__ _Float16 Pl[QT_ * 16 * 48];   // P tiles 16x32, stride 48 halves

  const int lane = threadIdx.x;
  const int l15  = lane & 15;
  const int hi   = lane >> 4;
  const int b    = blockIdx.z;
  const int h    = blockIdx.y;
  const int q0   = blockIdx.x * (16 * QT_);
  const int bh   = b * NH_ + h;

  const _Float16* Qh    = Q  + (size_t)bh * S_ * DK_;
  const _Float16* Kh    = K  + (size_t)bh * S_ * DK_;
  const _Float16* Vth   = Vt + (size_t)bh * DK_ * S_;
  const int*      maskB = mask + (size_t)b * S_ * S_;

  // Q fragments (A operand) per q-tile, dk split into two 32-chunks
  v16h aq[QT_][2];
#pragma unroll
  for (int qt = 0; qt < QT_; ++qt) {
#pragma unroll
    for (int c = 0; c < 2; ++c) {
      const _Float16* qp = Qh + (size_t)(q0 + 16 * qt + l15) * DK_ + 32 * c;
      aq[qt][c] = cat8(*(const v8h*)(qp + 8 * hi),
                       *(const v8h*)(qp + 16 + 8 * hi));
    }
  }

  float mrow[QT_][8], lrow[QT_][8];
  v8f acc[QT_][4];
#pragma unroll
  for (int qt = 0; qt < QT_; ++qt) {
#pragma unroll
    for (int r = 0; r < 8; ++r) { mrow[qt][r] = -__builtin_inff(); lrow[qt][r] = 0.0f; }
#pragma unroll
    for (int j = 0; j < 4; ++j) acc[qt][j] = zero8();
  }

  for (int kb = 0; kb < S_; kb += 32) {
    // prefetch next 32-key chunk of K, V and mask (global_prefetch_b8)
    if (kb + 32 < S_) {
      __builtin_prefetch(Kh + (size_t)(kb + 32 + lane) * DK_, 0, 1);
      __builtin_prefetch(Vth + (size_t)lane * S_ + kb + 32, 0, 1);
      __builtin_prefetch(Vth + (size_t)(lane + 32) * S_ + kb + 32, 0, 1);
      __builtin_prefetch(maskB + (size_t)(q0 + lane) * S_ + kb + 32, 0, 1);
    }

#pragma unroll
    for (int qt = 0; qt < QT_; ++qt) {
      const int qb = q0 + 16 * qt;
      // ---- scores: two 16-key tiles, each accumulated over dk = 2 x 32
      v8f sc[2];
#pragma unroll
      for (int t = 0; t < 2; ++t) {
        const _Float16* kp = Kh + (size_t)(kb + 16 * t + l15) * DK_;
        v8f z = zero8();
#pragma unroll
        for (int c = 0; c < 2; ++c) {
          const _Float16* kpc = kp + 32 * c;
          const v16h bk = cat8(*(const v8h*)(kpc + 8 * hi), *(const v8h*)(kpc + 16 + 8 * hi));
          z = __builtin_amdgcn_wmma_f32_16x16x32_f16(
              false, aq[qt][c], false, bk, (short)0, z, false, false);
        }
        sc[t] = z;
      }
      // ---- masked_fill(mask, -1e9)
#pragma unroll
      for (int t = 0; t < 2; ++t) {
#pragma unroll
        for (int r = 0; r < 8; ++r) {
          const int qq  = qb + r + (hi << 3);
          const int key = kb + 16 * t + l15;
          if (maskB[(size_t)qq * S_ + key] != 0) sc[t][r] = NEGV;
        }
      }
      // ---- online softmax (reduce across the 16-lane group holding one row)
      float f[8];
#pragma unroll
      for (int r = 0; r < 8; ++r) {
        float tmax = fmaxf(sc[0][r], sc[1][r]);
#pragma unroll
        for (int off = 1; off < 16; off <<= 1)
          tmax = fmaxf(tmax, __shfl_xor(tmax, off, 16));
        const float mnew = fmaxf(mrow[qt][r], tmax);
        f[r]        = __expf(mrow[qt][r] - mnew);
        mrow[qt][r] = mnew;
        const float p0 = __expf(sc[0][r] - mnew);
        const float p1 = __expf(sc[1][r] - mnew);
        sc[0][r] = p0; sc[1][r] = p1;
        float rs = p0 + p1;
#pragma unroll
        for (int off = 1; off < 16; off <<= 1)
          rs += __shfl_xor(rs, off, 16);
        lrow[qt][r] = lrow[qt][r] * f[r] + rs;
      }
#pragma unroll
      for (int j = 0; j < 4; ++j)
#pragma unroll
        for (int r = 0; r < 8; ++r) acc[qt][j][r] *= f[r];

      // ---- P tile: C layout -> LDS row-major 16x32
#pragma unroll
      for (int r = 0; r < 8; ++r) {
        Pl[(qt * 16 + r + (hi << 3)) * 48 + l15]      = (_Float16)sc[0][r];
        Pl[(qt * 16 + r + (hi << 3)) * 48 + 16 + l15] = (_Float16)sc[1][r];
      }
    }
    __syncthreads();
    v16h pf[QT_];
#pragma unroll
    for (int qt = 0; qt < QT_; ++qt) {
      const _Float16* pp = &Pl[(qt * 16 + l15) * 48];
      pf[qt] = cat8(*(const v8h*)(pp + 8 * hi),
                    *(const v8h*)(pp + 16 + 8 * hi));
    }
    __syncthreads();

    // ---- acc += P(16x32) x V(32x16); V fragment shared across both q-tiles
#pragma unroll
    for (int j = 0; j < 4; ++j) {
      const _Float16* vp = Vth + (size_t)((j << 4) + l15) * S_ + kb;
      const v16h bv = cat8(*(const v8h*)(vp + 8 * hi), *(const v8h*)(vp + 16 + 8 * hi));
#pragma unroll
      for (int qt = 0; qt < QT_; ++qt)
        acc[qt][j] = __builtin_amdgcn_wmma_f32_16x16x32_f16(
            false, pf[qt], false, bv, (short)0, acc[qt][j], false, false);
    }
  }

  // ---- normalize and store O as f16 [b, s, h*64 + dk]
#pragma unroll
  for (int qt = 0; qt < QT_; ++qt) {
#pragma unroll
    for (int r = 0; r < 8; ++r) {
      const float inv = 1.0f / lrow[qt][r];
      const int qq = q0 + 16 * qt + r + (hi << 3);
#pragma unroll
      for (int j = 0; j < 4; ++j) {
        const int dk = (j << 4) + l15;
        O[((size_t)(b * S_ + qq)) * HID_ + h * DK_ + dk] =
            (_Float16)(acc[qt][j][r] * inv);
      }
    }
  }
}

// ---------------------------------------------------------------------------
extern "C" void kernel_launch(void* const* d_in, const int* in_sizes, int n_in,
                              void* d_out, int out_size, void* d_ws, size_t ws_size,
                              hipStream_t stream) {
  (void)in_sizes; (void)n_in; (void)out_size; (void)ws_size;
  const float* q    = (const float*)d_in[0];
  const float* k    = (const float*)d_in[1];
  const float* v    = (const float*)d_in[2];
  const int*   mask = (const int*)d_in[3];
  const float* Wq   = (const float*)d_in[4];
  const float* Wk   = (const float*)d_in[5];
  const float* Wv   = (const float*)d_in[6];
  const float* Wo   = (const float*)d_in[7];

  // workspace: 4 x (4096*1024 f16) = 32 MB
  _Float16* qbuf = (_Float16*)d_ws;
  _Float16* kbuf = qbuf + (size_t)M_TOT * HID_;
  _Float16* vbuf = kbuf + (size_t)M_TOT * HID_;
  _Float16* obuf = vbuf + (size_t)M_TOT * HID_;

  dim3 ggrid(HID_ / 64, M_TOT / 64);
  gemm_mha<false><<<ggrid, 128, 0, stream>>>(q, Wq, qbuf, MODE_HEADS,   SCALE_);
  gemm_mha<false><<<ggrid, 128, 0, stream>>>(k, Wk, kbuf, MODE_HEADS,   1.0f);
  gemm_mha<false><<<ggrid, 128, 0, stream>>>(v, Wv, vbuf, MODE_HEADS_T, 1.0f);

  dim3 agrid(S_ / (16 * QT_), NH_, B_);
  attn_mha<<<agrid, 32, 0, stream>>>(qbuf, kbuf, vbuf, mask, obuf);

  gemm_mha<true><<<ggrid, 128, 0, stream>>>(obuf, Wo, d_out, MODE_F32, 1.0f);
}
